// MoEBlock_88029649699655
// MI455X (gfx1250) — compile-verified, matrix operand order checked by reading, not run on hardware
//
#include <hip/hip_runtime.h>
#include <hip/hip_bf16.h>
#include <math.h>

// ---------------------------------------------------------------------------
// MoE transformer block for MI455X (gfx1250):
//   wave32 + v_wmma_f32_16x16x32_bf16 + global_load_async_to_lds_b128 pipeline
// ---------------------------------------------------------------------------

typedef __attribute__((ext_vector_type(16))) __bf16 v16bf;
typedef __attribute__((ext_vector_type(8)))  __bf16 v8bf;
typedef __attribute__((ext_vector_type(8)))  float  v8f;

#define D_MODEL 1024
#define N_HEADS 16
#define HEAD_DIM 64
#define FFN_DIM 4096
#define N_EXPERTS 8
#define BATCH 4
#define SEQ 1024
#define NTOK (BATCH * SEQ)

// Assemble a 16x32 (or 32x16) bf16 WMMA fragment from LDS.
// Per ISA layout (16-bit A/B 16x32): lane L (m/n = L&15, kg = L>>4):
//   elems 0..7  <- K = kg*8 + 0..7        (16 contiguous bytes)
//   elems 8..15 <- K = 16 + kg*8 + 0..7   (16 contiguous bytes)
__device__ __forceinline__ v16bf load_frag(const __bf16* p0, const __bf16* p1) {
    v8bf lo = *(const v8bf*)p0;
    v8bf hi = *(const v8bf*)p1;
    return __builtin_shufflevector(lo, hi, 0, 1, 2, 3, 4, 5, 6, 7,
                                           8, 9, 10, 11, 12, 13, 14, 15);
}

__device__ __forceinline__ __bf16 f2bf(float f) { return (__bf16)f; }

__device__ __forceinline__ unsigned lds_off(const void* p) {
    return (unsigned)(uintptr_t)p;   // low 32 bits of generic shared addr = LDS offset
}

// CDNA5 async copy: 16B/lane global -> LDS, tracked by ASYNCcnt.
__device__ __forceinline__ void async_load_b128(unsigned ldsAddr, const void* gaddr) {
    asm volatile("global_load_async_to_lds_b128 %0, %1, off"
                 :: "v"(ldsAddr), "v"((unsigned long long)(uintptr_t)gaddr)
                 : "memory");
}
__device__ __forceinline__ void wait_async_le4() {
    asm volatile("s_wait_asynccnt 0x4" ::: "memory");
}
__device__ __forceinline__ void wait_async_0() {
    asm volatile("s_wait_asynccnt 0x0" ::: "memory");
}

// ---------------------------------------------------------------------------
// Async double-buffered bf16 GEMM (expert FFN path):
//   C[M,N] (+)= act( gatherA(A)[M,K] * B^T + bias ),  A:[M,K] bf16, B:[N,K] bf16
//   GELU_STORE_BF16: gelu epilogue, store bf16 (fc1 -> h)
//   SCATTER_ADD:     Cf[outRows[m]] += v * outScale[m] (fc2 combine, f32)
// Block: 128 threads (4 waves, 2x2), tile 64x64, BK=32, 2-deep LDS pipeline.
// ---------------------------------------------------------------------------
template <bool GELU_STORE_BF16, bool SCATTER_ADD>
__global__ __launch_bounds__(128)
void gemm_async_kernel(const __bf16* __restrict__ A, const __bf16* __restrict__ Bm,
                       const float* __restrict__ bias,
                       float* __restrict__ Cf, __bf16* __restrict__ Cbf,
                       const int* __restrict__ rowList,   // gather A rows (nullable)
                       const int* __restrict__ cntPtr,    // effective M (nullable)
                       const int* __restrict__ outRows,   // scatter C rows (nullable)
                       const float* __restrict__ outScale,// per-row scale (nullable)
                       int M, int N, int K) {
    const int tid  = threadIdx.x;
    const int lane = tid & 31;
    const int wave = tid >> 5;
    const int wr   = wave >> 1;
    const int wc   = wave & 1;
    const int lg   = lane >> 4;
    const int ln   = lane & 15;
    const int mBlock = blockIdx.y * 64;
    const int nBlock = blockIdx.x * 64;

    const int Meff = cntPtr ? *cntPtr : M;
    if (mBlock >= Meff) return;

    __shared__ __align__(16) __bf16 As[2][64 * 40];   // [m][k], stride-40 pad
    __shared__ __align__(16) __bf16 Bs[2][64 * 40];   // [n][k], stride-40 pad

    // Each thread owns 2 A-chunks + 2 B-chunks (chunk = 8 bf16 = 16B).
    // Tile = 64 rows x 4 chunks = 256 chunks; chunk c: row = c>>2, seg = c&3.
    const __bf16* aPtr[2];
    const __bf16* bPtr[2];
    unsigned aOff[2], bOff[2];
    for (int q = 0; q < 2; ++q) {
        const int c   = tid + q * 128;
        const int r   = c >> 2, seg = c & 3;
        int gr = mBlock + r;
        if (gr >= Meff) gr = Meff - 1;          // clamp: pad rows feed skipped C rows
        const int ar = rowList ? rowList[gr] : gr;
        aPtr[q] = A  + (long)ar * K + seg * 8;
        bPtr[q] = Bm + (long)(nBlock + r) * K + seg * 8;
        aOff[q] = (unsigned)((r * 40 + seg * 8) * 2);
        bOff[q] = (unsigned)((r * 40 + seg * 8) * 2);
    }
    const unsigned asBase[2] = { lds_off(&As[0][0]), lds_off(&As[1][0]) };
    const unsigned bsBase[2] = { lds_off(&Bs[0][0]), lds_off(&Bs[1][0]) };

    auto issue = [&](int kt, int buf) {
        const long k0 = (long)kt * 32;
        for (int q = 0; q < 2; ++q) {
            async_load_b128(asBase[buf] + aOff[q], aPtr[q] + k0);
            async_load_b128(bsBase[buf] + bOff[q], bPtr[q] + k0);
        }
    };

    v8f acc[2][2] = {};
    const int nk = K >> 5;
    issue(0, 0);
    for (int kt = 0; kt < nk; ++kt) {
        const int buf = kt & 1;
        const bool more = (kt + 1) < nk;
        if (more) issue(kt + 1, buf ^ 1);       // prefetch next tile into other buffer
        if (more) wait_async_le4();             // retire current tile (4 in flight = next)
        else      wait_async_0();
        __syncthreads();

        v16bf af[2], bfv[2];
        for (int i = 0; i < 2; ++i) {
            const int m = wr * 32 + i * 16 + ln;
            af[i] = load_frag(&As[buf][m * 40 + lg * 8], &As[buf][m * 40 + 16 + lg * 8]);
            const int n = wc * 32 + i * 16 + ln;
            bfv[i] = load_frag(&Bs[buf][n * 40 + lg * 8], &Bs[buf][n * 40 + 16 + lg * 8]);
        }
        for (int i = 0; i < 2; ++i)
            for (int j = 0; j < 2; ++j)
                acc[i][j] = __builtin_amdgcn_wmma_f32_16x16x32_bf16(
                    false, af[i], false, bfv[j], (short)0, acc[i][j], false, false);
        __syncthreads();                         // buffer reuse fence
    }

    for (int i = 0; i < 2; ++i) {
        for (int j = 0; j < 2; ++j) {
            const int nCol = nBlock + wc * 32 + j * 16 + ln;
            for (int r = 0; r < 8; ++r) {
                const int mRow = mBlock + wr * 32 + i * 16 + r + lg * 8;
                if (mRow >= Meff) continue;
                float v = acc[i][j][r];
                if (bias) v += bias[nCol];
                if (GELU_STORE_BF16) {
                    v = 0.5f * v * (1.0f + erff(v * 0.70710678118f));
                    Cbf[(long)mRow * N + nCol] = f2bf(v);
                } else if (SCATTER_ADD) {
                    const int orow = outRows ? outRows[mRow] : mRow;
                    const float s = outScale ? outScale[mRow] : 1.0f;
                    Cf[(long)orow * N + nCol] += v * s;   // experts serialized on stream
                } else {
                    Cf[(long)mRow * N + nCol] = v;
                }
            }
        }
    }
}

// ---------------------------------------------------------------------------
// f32-input WMMA GEMM for Wqkv/Wo ([K,N] f32 weights, transpose+convert on
// stage). C = A*B + bias (+res). Same 64x64x32 tiling.
// ---------------------------------------------------------------------------
__global__ __launch_bounds__(128)
void gemm_kn_kernel(const float* __restrict__ A, const float* __restrict__ Bm,
                    const float* __restrict__ bias, const float* __restrict__ res,
                    float* __restrict__ C, int M, int N, int K) {
    const int tid  = threadIdx.x;
    const int lane = tid & 31;
    const int wave = tid >> 5;
    const int wr = wave >> 1, wc = wave & 1;
    const int lg = lane >> 4, ln = lane & 15;
    const int mBlock = blockIdx.y * 64;
    const int nBlock = blockIdx.x * 64;

    __shared__ __align__(16) __bf16 As[64 * 40];
    __shared__ __align__(16) __bf16 Bs[64 * 40];

    v8f acc[2][2] = {};
    for (int k0 = 0; k0 < K; k0 += 32) {
        for (int e = tid; e < 64 * 32; e += 128) {
            const int r = e >> 5, c = e & 31;
            As[r * 40 + c] = f2bf(A[(long)(mBlock + r) * K + (k0 + c)]);
        }
        for (int e = tid; e < 64 * 32; e += 128) {
            const int kk = e >> 6, n = e & 63;   // n fastest: coalesced read
            Bs[n * 40 + kk] = f2bf(Bm[(long)(k0 + kk) * N + (nBlock + n)]);
        }
        if (k0 + 32 < K)
            __builtin_prefetch(&Bm[(long)(k0 + 32 + (tid >> 6)) * N + nBlock + (tid & 63)], 0, 0);
        __syncthreads();

        v16bf af[2], bfv[2];
        for (int i = 0; i < 2; ++i) {
            const int m = wr * 32 + i * 16 + ln;
            af[i] = load_frag(&As[m * 40 + lg * 8], &As[m * 40 + 16 + lg * 8]);
            const int n = wc * 32 + i * 16 + ln;
            bfv[i] = load_frag(&Bs[n * 40 + lg * 8], &Bs[n * 40 + 16 + lg * 8]);
        }
        for (int i = 0; i < 2; ++i)
            for (int j = 0; j < 2; ++j)
                acc[i][j] = __builtin_amdgcn_wmma_f32_16x16x32_bf16(
                    false, af[i], false, bfv[j], (short)0, acc[i][j], false, false);
        __syncthreads();
    }
    for (int i = 0; i < 2; ++i) {
        for (int j = 0; j < 2; ++j) {
            const int nCol = nBlock + wc * 32 + j * 16 + ln;
            for (int r = 0; r < 8; ++r) {
                const int mRow = mBlock + wr * 32 + i * 16 + r + lg * 8;
                float v = acc[i][j][r] + (bias ? bias[nCol] : 0.0f);
                if (res) v += res[(long)mRow * N + nCol];
                C[(long)mRow * N + nCol] = v;
            }
        }
    }
}

// ---------------------------------------------------------------------------
// LayerNorm (templated output type): one 256-thread block per token.
// ---------------------------------------------------------------------------
template <typename OutT>
__global__ __launch_bounds__(256)
void ln_kernel(const float* __restrict__ x, const float* __restrict__ w,
               const float* __restrict__ b, OutT* __restrict__ out) {
    __shared__ float red[256];
    const int t = blockIdx.x;
    const float* row = x + (long)t * D_MODEL;
    float s = 0.0f;
    for (int i = threadIdx.x; i < D_MODEL; i += 256) s += row[i];
    red[threadIdx.x] = s; __syncthreads();
    for (int o = 128; o > 0; o >>= 1) {
        if (threadIdx.x < o) red[threadIdx.x] += red[threadIdx.x + o];
        __syncthreads();
    }
    const float mu = red[0] * (1.0f / D_MODEL);
    __syncthreads();
    float vs = 0.0f;
    for (int i = threadIdx.x; i < D_MODEL; i += 256) {
        const float d = row[i] - mu; vs += d * d;
    }
    red[threadIdx.x] = vs; __syncthreads();
    for (int o = 128; o > 0; o >>= 1) {
        if (threadIdx.x < o) red[threadIdx.x] += red[threadIdx.x + o];
        __syncthreads();
    }
    const float rstd = rsqrtf(red[0] * (1.0f / D_MODEL) + 1e-5f);
    for (int i = threadIdx.x; i < D_MODEL; i += 256)
        out[(long)t * D_MODEL + i] = (OutT)((row[i] - mu) * rstd * w[i] + b[i]);
}

// ---------------------------------------------------------------------------
// f32 -> bf16 weight conversion (grid-stride).
// ---------------------------------------------------------------------------
__global__ __launch_bounds__(256)
void conv_bf16_kernel(const float* __restrict__ in, __bf16* __restrict__ out, long n) {
    long i = (long)blockIdx.x * blockDim.x + threadIdx.x;
    const long stride = (long)gridDim.x * blockDim.x;
    for (; i < n; i += stride) out[i] = f2bf(in[i]);
}

// ---------------------------------------------------------------------------
// Flash-style causal attention: 1 wave per (b, h, 16-query tile).
// QK^T and P*V via bf16 WMMA; online softmax; 32-key tiles.
// ---------------------------------------------------------------------------
__global__ __launch_bounds__(32)
void attn_kernel(const float* __restrict__ qkv, float* __restrict__ attn) {
    const int lane = threadIdx.x;
    const int qt = blockIdx.x, h = blockIdx.y, b = blockIdx.z;
    const int lg = lane >> 4, ln = lane & 15;
    const long rowStride = 3 * D_MODEL;

    __shared__ __align__(16) __bf16 Qs[16 * 72];
    __shared__ __align__(16) __bf16 Ks[32 * 72];
    __shared__ __align__(16) __bf16 Vs[64 * 40];
    __shared__ __align__(16) __bf16 Ps[16 * 40];

    for (int e = lane; e < 16 * 64; e += 32) {
        const int r = e >> 6, c = e & 63;
        const long tok = (long)b * SEQ + qt * 16 + r;
        Qs[r * 72 + c] = f2bf(qkv[tok * rowStride + h * 64 + c]);
    }
    __syncthreads();
    v16bf aQ[2];
    for (int c = 0; c < 2; ++c)
        aQ[c] = load_frag(&Qs[ln * 72 + c * 32 + lg * 8],
                          &Qs[ln * 72 + c * 32 + 16 + lg * 8]);

    float mrow[8], lrow[8];
    v8f O[4] = {};
    for (int r = 0; r < 8; ++r) { mrow[r] = -1e30f; lrow[r] = 0.0f; }

    const int nkt = qt / 2 + 1;
    for (int kt = 0; kt < nkt; ++kt) {
        __syncthreads();
        for (int e = lane; e < 32 * 64; e += 32) {
            const int r = e >> 6, c = e & 63;
            const long tok = (long)b * SEQ + kt * 32 + r;
            Ks[r * 72 + c] = f2bf(qkv[tok * rowStride + 1024 + h * 64 + c]);
            Vs[c * 40 + r] = f2bf(qkv[tok * rowStride + 2048 + h * 64 + c]);
        }
        __syncthreads();

        v8f sf[2];
        for (int nh = 0; nh < 2; ++nh) {
            v8f s = {};
            const int n = nh * 16 + ln;
            for (int c = 0; c < 2; ++c) {
                v16bf bK = load_frag(&Ks[n * 72 + c * 32 + lg * 8],
                                     &Ks[n * 72 + c * 32 + 16 + lg * 8]);
                s = __builtin_amdgcn_wmma_f32_16x16x32_bf16(
                        false, aQ[c], false, bK, (short)0, s, false, false);
            }
            sf[nh] = s;
        }

        float p0v[8], p1v[8], newm[8];
        for (int r = 0; r < 8; ++r) {
            const int m = r + lg * 8;
            const int qg = qt * 16 + m;
            float v0 = sf[0][r] * 0.125f;
            float v1 = sf[1][r] * 0.125f;
            if (kt * 32 + ln > qg)      v0 = -1e30f;
            if (kt * 32 + 16 + ln > qg) v1 = -1e30f;
            p0v[r] = v0; p1v[r] = v1;
            float mx = fmaxf(v0, v1);
            for (int off = 1; off < 16; off <<= 1)
                mx = fmaxf(mx, __shfl_xor(mx, off, 32));
            newm[r] = fmaxf(mrow[r], mx);
        }
        for (int r = 0; r < 8; ++r) {
            const float resc = expf(mrow[r] - newm[r]);
            const float e0 = expf(p0v[r] - newm[r]);
            const float e1 = expf(p1v[r] - newm[r]);
            float rs = e0 + e1;
            for (int off = 1; off < 16; off <<= 1)
                rs += __shfl_xor(rs, off, 32);
            lrow[r] = lrow[r] * resc + rs;
            mrow[r] = newm[r];
            for (int d = 0; d < 4; ++d) O[d][r] *= resc;
            const int m = r + lg * 8;
            Ps[m * 40 + ln]      = f2bf(e0);
            Ps[m * 40 + 16 + ln] = f2bf(e1);
        }
        __syncthreads();

        v16bf aP = load_frag(&Ps[ln * 40 + lg * 8], &Ps[ln * 40 + 16 + lg * 8]);
        for (int d = 0; d < 4; ++d) {
            const int dn = d * 16 + ln;
            v16bf bV = load_frag(&Vs[dn * 40 + lg * 8], &Vs[dn * 40 + 16 + lg * 8]);
            O[d] = __builtin_amdgcn_wmma_f32_16x16x32_bf16(
                       false, aP, false, bV, (short)0, O[d], false, false);
        }
    }

    for (int r = 0; r < 8; ++r) {
        const float inv = 1.0f / lrow[r];
        const int m = r + lg * 8;
        const long tok = (long)b * SEQ + qt * 16 + m;
        for (int d = 0; d < 4; ++d)
            attn[tok * D_MODEL + h * 64 + d * 16 + ln] = O[d][r] * inv;
    }
}

// ---------------------------------------------------------------------------
// Gate: wave per token over bf16 nx2. Top-2, softmax over selected, append
// to per-expert compacted token lists.
// ---------------------------------------------------------------------------
__global__ __launch_bounds__(32)
void gate_kernel(const __bf16* __restrict__ nx2, const float* __restrict__ gW,
                 int* __restrict__ cnt, int* __restrict__ elist,
                 float* __restrict__ egw) {
    const int t = blockIdx.x, lane = threadIdx.x;
    const __bf16* row = nx2 + (long)t * D_MODEL;
    float acc[N_EXPERTS] = {};
    for (int d = lane; d < D_MODEL; d += 32) {
        const float xv = (float)row[d];
        const float* g = gW + (long)d * N_EXPERTS;
        for (int e = 0; e < N_EXPERTS; ++e) acc[e] += xv * g[e];
    }
    for (int e = 0; e < N_EXPERTS; ++e)
        for (int off = 1; off < 32; off <<= 1)
            acc[e] += __shfl_xor(acc[e], off, 32);
    if (lane == 0) {
        int e0 = 0; float v0 = acc[0];
        for (int e = 1; e < N_EXPERTS; ++e)
            if (acc[e] > v0) { v0 = acc[e]; e0 = e; }
        int e1 = -1; float v1 = -3e38f;
        for (int e = 0; e < N_EXPERTS; ++e)
            if (e != e0 && acc[e] > v1) { v1 = acc[e]; e1 = e; }
        const float m = fmaxf(v0, v1);
        float w0 = expf(v0 - m), w1 = expf(v1 - m);
        const float inv = 1.0f / (w0 + w1);
        w0 *= inv; w1 *= inv;
        int p = atomicAdd(&cnt[e0], 1);
        elist[e0 * NTOK + p] = t; egw[e0 * NTOK + p] = w0;
        p = atomicAdd(&cnt[e1], 1);
        elist[e1 * NTOK + p] = t; egw[e1 * NTOK + p] = w1;
    }
}

__global__ void zero_cnt_kernel(int* __restrict__ cnt) {
    if (threadIdx.x < N_EXPERTS) cnt[threadIdx.x] = 0;
}

// ---------------------------------------------------------------------------
// Host-side orchestration (all on `stream`, graph-capture safe).
// ---------------------------------------------------------------------------
extern "C" void kernel_launch(void* const* d_in, const int* in_sizes, int n_in,
                              void* d_out, int out_size, void* d_ws, size_t ws_size,
                              hipStream_t stream) {
    (void)in_sizes; (void)n_in; (void)out_size; (void)ws_size;

    const float* x      = (const float*)d_in[0];
    const float* ln1_w  = (const float*)d_in[1];
    const float* ln1_b  = (const float*)d_in[2];
    const float* ln2_w  = (const float*)d_in[3];
    const float* ln2_b  = (const float*)d_in[4];
    const float* Wqkv   = (const float*)d_in[5];
    const float* bqkv   = (const float*)d_in[6];
    const float* Wo     = (const float*)d_in[7];
    const float* bo     = (const float*)d_in[8];
    const float* gate_W = (const float*)d_in[9];
    const float* fc1_w  = (const float*)d_in[10];
    const float* fc1_b  = (const float*)d_in[11];
    const float* fc2_w  = (const float*)d_in[12];
    const float* fc2_b  = (const float*)d_in[13];
    float* out = (float*)d_out;

    // workspace layout
    char* ws = (char*)d_ws;
    float*  nx     = (float*)ws;   ws += (size_t)NTOK * D_MODEL * 4;            // 16 MB
    float*  qkv    = (float*)ws;   ws += (size_t)NTOK * 3 * D_MODEL * 4;        // 48 MB
    float*  attn   = (float*)ws;   ws += (size_t)NTOK * D_MODEL * 4;            // 16 MB
    __bf16* nx2bf  = (__bf16*)ws;  ws += (size_t)NTOK * D_MODEL * 2;            //  8 MB
    __bf16* hbf    = (__bf16*)ws;  ws += (size_t)NTOK * FFN_DIM * 2;            // 32 MB
    __bf16* fc1wbf = (__bf16*)ws;  ws += (size_t)N_EXPERTS * FFN_DIM * D_MODEL * 2; // 67 MB
    __bf16* fc2wbf = (__bf16*)ws;  ws += (size_t)N_EXPERTS * D_MODEL * FFN_DIM * 2; // 67 MB
    int*    elist  = (int*)ws;     ws += (size_t)N_EXPERTS * NTOK * 4;
    float*  egw    = (float*)ws;   ws += (size_t)N_EXPERTS * NTOK * 4;
    int*    cnt    = (int*)ws;     ws += 256;

    const long wElems = (long)N_EXPERTS * FFN_DIM * D_MODEL;

    // 0. Expert weights -> bf16 (already [N][K]); enables async-DMA GEMM path.
    conv_bf16_kernel<<<8192, 256, 0, stream>>>(fc1_w, fc1wbf, wElems);
    conv_bf16_kernel<<<8192, 256, 0, stream>>>(fc2_w, fc2wbf, wElems);

    // 1. LN1 (f32 out)
    ln_kernel<float><<<NTOK, 256, 0, stream>>>(x, ln1_w, ln1_b, nx);

    // 2. QKV GEMM: [4096,1024] x Wqkv[1024,3072] + bqkv
    gemm_kn_kernel<<<dim3(3 * D_MODEL / 64, NTOK / 64), 128, 0, stream>>>(
        nx, Wqkv, bqkv, nullptr, qkv, NTOK, 3 * D_MODEL, D_MODEL);

    // 3. Causal flash attention (WMMA)
    attn_kernel<<<dim3(SEQ / 16, N_HEADS, BATCH), 32, 0, stream>>>(qkv, attn);

    // 4. Output projection + residual: out = x + attn @ Wo + bo
    gemm_kn_kernel<<<dim3(D_MODEL / 64, NTOK / 64), 128, 0, stream>>>(
        attn, Wo, bo, x, out, NTOK, D_MODEL, D_MODEL);

    // 5. LN2 -> bf16
    ln_kernel<__bf16><<<NTOK, 256, 0, stream>>>(out, ln2_w, ln2_b, nx2bf);

    // 6. Gate: top-2 routing into compacted per-expert lists
    zero_cnt_kernel<<<1, 32, 0, stream>>>(cnt);
    gate_kernel<<<NTOK, 32, 0, stream>>>(nx2bf, gate_W, cnt, elist, egw);

    // 7. Per-expert FFN via async double-buffered WMMA GEMMs (stream-serialized)
    for (int e = 0; e < N_EXPERTS; ++e) {
        // fc1: h = gelu(nx2[list] @ fc1_w[e]^T + fc1_b[e])  -> bf16
        gemm_async_kernel<true, false><<<dim3(FFN_DIM / 64, NTOK / 64), 128, 0, stream>>>(
            nx2bf, fc1wbf + (size_t)e * FFN_DIM * D_MODEL,
            fc1_b + (size_t)e * FFN_DIM,
            nullptr, hbf,
            elist + (size_t)e * NTOK, cnt + e, nullptr, nullptr,
            NTOK, FFN_DIM, D_MODEL);
        // fc2: out[list[m]] += (h @ fc2_w[e]^T + fc2_b[e]) * gate_w[m]
        gemm_async_kernel<false, true><<<dim3(D_MODEL / 64, NTOK / 64), 128, 0, stream>>>(
            hbf, fc2wbf + (size_t)e * D_MODEL * FFN_DIM,
            fc2_b + (size_t)e * D_MODEL,
            out, nullptr,
            nullptr, cnt + e, elist + (size_t)e * NTOK, egw + (size_t)e * NTOK,
            NTOK, D_MODEL, FFN_DIM);
    }
}
